// MultiHeadDiffAttention_1348619731552
// MI455X (gfx1250) — compile-verified, matrix operand order checked by reading, not run on hardware
//
#include <hip/hip_runtime.h>
#include <hip/hip_bf16.h>

typedef _Float16 f16;
typedef _Float16 v16h __attribute__((ext_vector_type(16)));
typedef _Float16 h8   __attribute__((ext_vector_type(8)));
typedef float    v8f  __attribute__((ext_vector_type(8)));

union F16Frag { v16h v; h8 h[2]; };

constexpr int Bn = 4, Tn = 1024, Cn = 768, Hn = 12, HSn = 64, VDn = 128;
constexpr int HVDn = Hn * VDn;          // 1536
constexpr int Mn = Bn * Tn;             // 4096

// ---------------- workspace layout (bytes, all chunks 256B-aligned) ----------
constexpr size_t OFF_XH   = 0;                                      // x f16 [M,C]
constexpr size_t OFF_WQ1T = OFF_XH   + (size_t)Mn * Cn * 2;
constexpr size_t OFF_WK1T = OFF_WQ1T + (size_t)Hn * HSn * Cn * 2;   // W [H,HS,C]
constexpr size_t OFF_WQ2T = OFF_WK1T + (size_t)Hn * HSn * Cn * 2;
constexpr size_t OFF_WK2T = OFF_WQ2T + (size_t)Hn * HSn * Cn * 2;
constexpr size_t OFF_WVT  = OFF_WK2T + (size_t)Hn * HSn * Cn * 2;   // [H,VD,C]
constexpr size_t OFF_WPH  = OFF_WVT  + (size_t)Hn * VDn * Cn * 2;   // Wp f16 [C,HVD]
constexpr size_t OFF_Q1   = OFF_WPH  + (size_t)Cn * HVDn * 2;       // [B,H,T,HS]
constexpr size_t OFF_K1   = OFF_Q1   + (size_t)Bn * Hn * Tn * HSn * 2;
constexpr size_t OFF_Q2   = OFF_K1   + (size_t)Bn * Hn * Tn * HSn * 2;
constexpr size_t OFF_K2   = OFF_Q2   + (size_t)Bn * Hn * Tn * HSn * 2;
constexpr size_t OFF_VT   = OFF_K2   + (size_t)Bn * Hn * Tn * HSn * 2; // [B,H,VD,T]
constexpr size_t OFF_ATT  = OFF_VT   + (size_t)Bn * Hn * VDn * Tn * 2; // f32 [B,T,HVD]
constexpr size_t OFF_Y    = OFF_ATT  + (size_t)Bn * Tn * HVDn * 4;     // f16 [M,HVD]
constexpr size_t OFF_LAM  = OFF_Y    + (size_t)Mn * HVDn * 2;          // f32 [H]

// ---------------- conversion kernels ----------------------------------------
__global__ void __launch_bounds__(256) cvt_f32_f16(const float* __restrict__ in,
                                                   f16* __restrict__ out, int n) {
  int i = blockIdx.x * 256 + threadIdx.x;
  if (i < n) out[i] = (f16)in[i];
}

// in [H,C,D] (fp32) -> out [H,D,C] (f16)
__global__ void __launch_bounds__(256) cvt_w_tr(const float* __restrict__ in,
                                                f16* __restrict__ out, int D, int total) {
  int i = blockIdx.x * 256 + threadIdx.x;
  if (i >= total) return;
  int d = i % D;
  int c = (i / D) % Cn;
  int h = i / (D * Cn);
  out[((size_t)h * D + d) * Cn + c] = (f16)in[i];
}

// ---------------- lambda kernel ----------------------------------------------
__global__ void lam_kernel(const float* __restrict__ lq1, const float* __restrict__ lk1,
                           const float* __restrict__ lq2, const float* __restrict__ lk2,
                           const int* __restrict__ lidx, float* __restrict__ lam) {
  int h = blockIdx.x, d = threadIdx.x;                       // 64 threads
  float v = __expf(lq1[h * HSn + d] * lk1[h * HSn + d]) -
            __expf(lq2[h * HSn + d] * lk2[h * HSn + d]);
  __shared__ float sred[2];
  for (int m = 16; m >= 1; m >>= 1) v += __shfl_xor(v, m, 32);
  if ((threadIdx.x & 31) == 0) sred[threadIdx.x >> 5] = v;
  __syncthreads();
  if (threadIdx.x == 0) {
    float li = (float)lidx[0];
    float dyn = 0.8f - 0.6f * __expf(-0.3f * (li - 1.0f));
    lam[h] = (sred[0] + sred[1]) / 64.0f + dyn;
  }
}

// ---------------- WMMA GEMM: D = A(MxK) * Bt(NxK)^T --------------------------
// MODE 0: out f16 [B,H,T,N]   (Q/K projections)
// MODE 1: out f16 [B,H,N,T]   (V projection, transposed store)
// MODE 2: out f32 [M,N] + bias (final projection)
template <int MODE>
__global__ void __launch_bounds__(256) gemm_wmma(const f16* __restrict__ A,
                                                 const f16* __restrict__ Bt,
                                                 void* __restrict__ outp,
                                                 const float* __restrict__ bias,
                                                 int K, int N) {
  const int lane = threadIdx.x & 31;
  const int wave = threadIdx.x >> 5;
  const int l15 = lane & 15, lh = lane >> 4;
  const int n0 = blockIdx.x * 16;
  const int m0 = blockIdx.y * 128 + wave * 16;
  const int h  = blockIdx.z;
  const f16* Ar = A + (size_t)(m0 + l15) * K;
  const f16* Br = Bt + (size_t)h * N * K + (size_t)(n0 + l15) * K;
  const int kofs = lh * 8;
  v8f acc = {0.f, 0.f, 0.f, 0.f, 0.f, 0.f, 0.f, 0.f};
  for (int k0 = 0; k0 < K; k0 += 32) {
    // prefetch next k-step one cacheline ahead (speculative, no counter cost)
    __builtin_prefetch(Ar + k0 + 64, 0, 1);
    __builtin_prefetch(Br + k0 + 64, 0, 1);
    F16Frag a, b;
    a.h[0] = *(const h8*)(Ar + k0 + kofs);
    a.h[1] = *(const h8*)(Ar + k0 + kofs + 16);
    b.h[0] = *(const h8*)(Br + k0 + kofs);
    b.h[1] = *(const h8*)(Br + k0 + kofs + 16);
    acc = __builtin_amdgcn_wmma_f32_16x16x32_f16(false, a.v, false, b.v,
                                                 (short)0, acc, false, false);
  }
  const int col = n0 + l15;
#pragma unroll
  for (int g = 0; g < 8; ++g) {
    int m = m0 + lh * 8 + g;
    if (MODE == 0) {
      int b = m >> 10, t = m & 1023;
      ((f16*)outp)[((size_t)(b * Hn + h) * Tn + t) * N + col] = (f16)acc[g];
    } else if (MODE == 1) {
      int b = m >> 10, t = m & 1023;
      ((f16*)outp)[((size_t)(b * Hn + h) * N + col) * Tn + t] = (f16)acc[g];
    } else {
      ((float*)outp)[(size_t)m * N + col] = acc[g] + bias[col];
    }
  }
}

// ---------------- differential flash attention -------------------------------
__global__ void __launch_bounds__(256) attn_kernel(
    const f16* __restrict__ q1h, const f16* __restrict__ k1h,
    const f16* __restrict__ q2h, const f16* __restrict__ k2h,
    const f16* __restrict__ vth, const float* __restrict__ lamp,
    float* __restrict__ att) {
  const int lane = threadIdx.x & 31, wave = threadIdx.x >> 5;
  const int l15 = lane & 15, lh = lane >> 4;
  const int bh = blockIdx.y, b = bh / Hn, h = bh % Hn;
  const int qrow0 = blockIdx.x * 128 + wave * 16;
  const int kofs = lh * 8;
  const size_t qk = (size_t)bh * Tn * HSn;
  const f16* q1p = q1h + qk; const f16* k1p = k1h + qk;
  const f16* q2p = q2h + qk; const f16* k2p = k2h + qk;
  const f16* vtp = vth + (size_t)bh * VDn * Tn;
  const float lam = lamp[h];

  __shared__ f16 ldsP[8][16 * 32];
  f16* lp = &ldsP[wave][0];

  // Q fragments (A-layout, head-dim split 0..31 / 32..63)
  F16Frag aq1lo, aq1hi, aq2lo, aq2hi;
  {
    const f16* r = q1p + (size_t)(qrow0 + l15) * HSn;
    aq1lo.h[0] = *(const h8*)(r + kofs);      aq1lo.h[1] = *(const h8*)(r + kofs + 16);
    aq1hi.h[0] = *(const h8*)(r + 32 + kofs); aq1hi.h[1] = *(const h8*)(r + 32 + kofs + 16);
  }
  {
    const f16* r = q2p + (size_t)(qrow0 + l15) * HSn;
    aq2lo.h[0] = *(const h8*)(r + kofs);      aq2lo.h[1] = *(const h8*)(r + kofs + 16);
    aq2hi.h[0] = *(const h8*)(r + 32 + kofs); aq2hi.h[1] = *(const h8*)(r + 32 + kofs + 16);
  }

  const v8f vzero = {0.f, 0.f, 0.f, 0.f, 0.f, 0.f, 0.f, 0.f};
  v8f O1[8], O2[8];
  float m1[8], m2[8], l1[8], l2[8];
#pragma unroll
  for (int j = 0; j < 8; ++j) { O1[j] = vzero; O2[j] = vzero; }
#pragma unroll
  for (int g = 0; g < 8; ++g) { m1[g] = -1e30f; m2[g] = -1e30f; l1[g] = 0.f; l2[g] = 0.f; }

  const int qmax = qrow0 + 15;
  for (int s0 = 0; s0 <= qmax; s0 += 32) {
    // prefetch next key block's K rows (speculative)
    __builtin_prefetch(k1p + (size_t)(s0 + 32 + l15) * HSn, 0, 1);
    __builtin_prefetch(k2p + (size_t)(s0 + 32 + l15) * HSn, 0, 1);
    v8f c1[2], c2[2];
#pragma unroll
    for (int tile = 0; tile < 2; ++tile) {
      const f16* r1 = k1p + (size_t)(s0 + tile * 16 + l15) * HSn;
      const f16* r2 = k2p + (size_t)(s0 + tile * 16 + l15) * HSn;
      F16Frag blo, bhi;
      blo.h[0] = *(const h8*)(r1 + kofs);      blo.h[1] = *(const h8*)(r1 + kofs + 16);
      bhi.h[0] = *(const h8*)(r1 + 32 + kofs); bhi.h[1] = *(const h8*)(r1 + 32 + kofs + 16);
      v8f c = vzero;
      c = __builtin_amdgcn_wmma_f32_16x16x32_f16(false, aq1lo.v, false, blo.v, (short)0, c, false, false);
      c = __builtin_amdgcn_wmma_f32_16x16x32_f16(false, aq1hi.v, false, bhi.v, (short)0, c, false, false);
      c1[tile] = c;
      blo.h[0] = *(const h8*)(r2 + kofs);      blo.h[1] = *(const h8*)(r2 + kofs + 16);
      bhi.h[0] = *(const h8*)(r2 + 32 + kofs); bhi.h[1] = *(const h8*)(r2 + 32 + kofs + 16);
      c = vzero;
      c = __builtin_amdgcn_wmma_f32_16x16x32_f16(false, aq2lo.v, false, blo.v, (short)0, c, false, false);
      c = __builtin_amdgcn_wmma_f32_16x16x32_f16(false, aq2hi.v, false, bhi.v, (short)0, c, false, false);
      c2[tile] = c;
    }
    // scale + causal mask (D-layout: row = lh*8+g, col = s0+tile*16+l15)
    const bool needmask = (s0 + 31 > qrow0);
#pragma unroll
    for (int tile = 0; tile < 2; ++tile)
#pragma unroll
      for (int g = 0; g < 8; ++g) {
        float x1 = c1[tile][g] * 0.125f, x2 = c2[tile][g] * 0.125f;
        if (needmask) {
          int q = qrow0 + lh * 8 + g, kk = s0 + tile * 16 + l15;
          if (kk > q) { x1 = -1e30f; x2 = -1e30f; }
        }
        c1[tile][g] = x1; c2[tile][g] = x2;
      }
    float corr1[8], corr2[8];
    // attn stream 1: online softmax, probs -> LDS (A-layout rebuild)
#pragma unroll
    for (int g = 0; g < 8; ++g) {
      float rmax = fmaxf(c1[0][g], c1[1][g]);
      for (int msk = 1; msk < 16; msk <<= 1) rmax = fmaxf(rmax, __shfl_xor(rmax, msk, 16));
      float mn = fmaxf(m1[g], rmax);
      corr1[g] = __expf(m1[g] - mn);
      float p0 = __expf(c1[0][g] - mn), p1 = __expf(c1[1][g] - mn);
      float rs = p0 + p1;
      for (int msk = 1; msk < 16; msk <<= 1) rs += __shfl_xor(rs, msk, 16);
      l1[g] = l1[g] * corr1[g] + rs; m1[g] = mn;
      lp[(lh * 8 + g) * 32 + l15] = (f16)p0;
      lp[(lh * 8 + g) * 32 + 16 + l15] = (f16)p1;
    }
    asm volatile("s_wait_dscnt 0" ::: "memory");
    F16Frag pa1;
    pa1.h[0] = *(const h8*)(lp + l15 * 32 + kofs);
    pa1.h[1] = *(const h8*)(lp + l15 * 32 + 16 + kofs);
    // attn stream 2
#pragma unroll
    for (int g = 0; g < 8; ++g) {
      float rmax = fmaxf(c2[0][g], c2[1][g]);
      for (int msk = 1; msk < 16; msk <<= 1) rmax = fmaxf(rmax, __shfl_xor(rmax, msk, 16));
      float mn = fmaxf(m2[g], rmax);
      corr2[g] = __expf(m2[g] - mn);
      float p0 = __expf(c2[0][g] - mn), p1 = __expf(c2[1][g] - mn);
      float rs = p0 + p1;
      for (int msk = 1; msk < 16; msk <<= 1) rs += __shfl_xor(rs, msk, 16);
      l2[g] = l2[g] * corr2[g] + rs; m2[g] = mn;
      lp[(lh * 8 + g) * 32 + l15] = (f16)p0;
      lp[(lh * 8 + g) * 32 + 16 + l15] = (f16)p1;
    }
    asm volatile("s_wait_dscnt 0" ::: "memory");
    F16Frag pa2;
    pa2.h[0] = *(const h8*)(lp + l15 * 32 + kofs);
    pa2.h[1] = *(const h8*)(lp + l15 * 32 + 16 + kofs);
    // rescale accumulators
#pragma unroll
    for (int j = 0; j < 8; ++j)
#pragma unroll
      for (int g = 0; g < 8; ++g) { O1[j][g] *= corr1[g]; O2[j][g] *= corr2[g]; }
    // O += P @ V   (V transposed [VD,T]: contiguous keys per lane)
#pragma unroll
    for (int j = 0; j < 8; ++j) {
      const f16* vr = vtp + (size_t)(j * 16 + l15) * Tn + s0;
      F16Frag bv;
      bv.h[0] = *(const h8*)(vr + kofs);
      bv.h[1] = *(const h8*)(vr + 16 + kofs);
      O1[j] = __builtin_amdgcn_wmma_f32_16x16x32_f16(false, pa1.v, false, bv.v, (short)0, O1[j], false, false);
      O2[j] = __builtin_amdgcn_wmma_f32_16x16x32_f16(false, pa2.v, false, bv.v, (short)0, O2[j], false, false);
    }
  }
  float inv1[8], inv2[8];
#pragma unroll
  for (int g = 0; g < 8; ++g) { inv1[g] = 1.f / l1[g]; inv2[g] = 1.f / l2[g]; }
#pragma unroll
  for (int j = 0; j < 8; ++j)
#pragma unroll
    for (int g = 0; g < 8; ++g) {
      int q = qrow0 + lh * 8 + g;
      att[((size_t)(b * Tn + q)) * HVDn + h * VDn + j * 16 + l15] =
          O1[j][g] * inv1[g] - lam * O2[j][g] * inv2[g];
    }
}

// ---------------- group layernorm (full-row LN) + 0.2 scale ------------------
__global__ void __launch_bounds__(256) gnorm_kernel(const float* __restrict__ att,
                                                    const float* __restrict__ gw,
                                                    const float* __restrict__ gb,
                                                    f16* __restrict__ y) {
  __shared__ float rs[256], rq[256];
  const int row = blockIdx.x, tid = threadIdx.x;
  const float* xr = att + (size_t)row * HVDn;
  float s = 0.f, q = 0.f;
#pragma unroll
  for (int k = 0; k < 6; ++k) { float v = xr[tid + k * 256]; s += v; q += v * v; }
  rs[tid] = s; rq[tid] = q; __syncthreads();
  for (int off = 128; off; off >>= 1) {
    if (tid < off) { rs[tid] += rs[tid + off]; rq[tid] += rq[tid + off]; }
    __syncthreads();
  }
  float mu = rs[0] * (1.f / HVDn);
  float var = rq[0] * (1.f / HVDn) - mu * mu;
  float rstd = rsqrtf(var + 1e-5f);
  f16* yr = y + (size_t)row * HVDn;
#pragma unroll
  for (int k = 0; k < 6; ++k) {
    int i = tid + k * 256;
    yr[i] = (f16)(((xr[i] - mu) * rstd * gw[i] + gb[i]) * 0.2f);
  }
}

// ---------------- launch ------------------------------------------------------
extern "C" void kernel_launch(void* const* d_in, const int* in_sizes, int n_in,
                              void* d_out, int out_size, void* d_ws, size_t ws_size,
                              hipStream_t stream) {
  (void)in_sizes; (void)n_in; (void)out_size; (void)ws_size;
  const float* x    = (const float*)d_in[0];
  const float* Wq1  = (const float*)d_in[1];
  const float* Wk1  = (const float*)d_in[2];
  const float* Wq2  = (const float*)d_in[3];
  const float* Wk2  = (const float*)d_in[4];
  const float* Wv   = (const float*)d_in[5];
  const float* lq1  = (const float*)d_in[6];
  const float* lk1  = (const float*)d_in[7];
  const float* lq2  = (const float*)d_in[8];
  const float* lk2  = (const float*)d_in[9];
  const float* gn_w = (const float*)d_in[10];
  const float* gn_b = (const float*)d_in[11];
  const float* Wp   = (const float*)d_in[12];
  const float* bp   = (const float*)d_in[13];
  const int*   lidx = (const int*)d_in[14];

  char* ws = (char*)d_ws;
  f16* xh    = (f16*)(ws + OFF_XH);
  f16* wq1t  = (f16*)(ws + OFF_WQ1T);
  f16* wk1t  = (f16*)(ws + OFF_WK1T);
  f16* wq2t  = (f16*)(ws + OFF_WQ2T);
  f16* wk2t  = (f16*)(ws + OFF_WK2T);
  f16* wvt   = (f16*)(ws + OFF_WVT);
  f16* wph   = (f16*)(ws + OFF_WPH);
  f16* q1b   = (f16*)(ws + OFF_Q1);
  f16* k1b   = (f16*)(ws + OFF_K1);
  f16* q2b   = (f16*)(ws + OFF_Q2);
  f16* k2b   = (f16*)(ws + OFF_K2);
  f16* vtb   = (f16*)(ws + OFF_VT);
  float* att = (float*)(ws + OFF_ATT);
  f16* yh    = (f16*)(ws + OFF_Y);
  float* lam = (float*)(ws + OFF_LAM);

  dim3 blk(256);
  cvt_f32_f16<<<(Mn * Cn + 255) / 256, blk, 0, stream>>>(x, xh, Mn * Cn);
  const int wqk_total = Hn * Cn * HSn;
  cvt_w_tr<<<(wqk_total + 255) / 256, blk, 0, stream>>>(Wq1, wq1t, HSn, wqk_total);
  cvt_w_tr<<<(wqk_total + 255) / 256, blk, 0, stream>>>(Wk1, wk1t, HSn, wqk_total);
  cvt_w_tr<<<(wqk_total + 255) / 256, blk, 0, stream>>>(Wq2, wq2t, HSn, wqk_total);
  cvt_w_tr<<<(wqk_total + 255) / 256, blk, 0, stream>>>(Wk2, wk2t, HSn, wqk_total);
  const int wv_total = Hn * Cn * VDn;
  cvt_w_tr<<<(wv_total + 255) / 256, blk, 0, stream>>>(Wv, wvt, VDn, wv_total);
  cvt_f32_f16<<<(Cn * HVDn + 255) / 256, blk, 0, stream>>>(Wp, wph, Cn * HVDn);
  lam_kernel<<<Hn, 64, 0, stream>>>(lq1, lk1, lq2, lk2, lidx, lam);

  gemm_wmma<0><<<dim3(HSn / 16, Mn / 128, Hn), blk, 0, stream>>>(xh, wq1t, q1b, nullptr, Cn, HSn);
  gemm_wmma<0><<<dim3(HSn / 16, Mn / 128, Hn), blk, 0, stream>>>(xh, wk1t, k1b, nullptr, Cn, HSn);
  gemm_wmma<0><<<dim3(HSn / 16, Mn / 128, Hn), blk, 0, stream>>>(xh, wq2t, q2b, nullptr, Cn, HSn);
  gemm_wmma<0><<<dim3(HSn / 16, Mn / 128, Hn), blk, 0, stream>>>(xh, wk2t, k2b, nullptr, Cn, HSn);
  gemm_wmma<1><<<dim3(VDn / 16, Mn / 128, Hn), blk, 0, stream>>>(xh, wvt, vtb, nullptr, Cn, VDn);

  attn_kernel<<<dim3(Tn / 128, Bn * Hn), blk, 0, stream>>>(q1b, k1b, q2b, k2b, vtb, lam, att);
  gnorm_kernel<<<Mn, blk, 0, stream>>>(att, gn_w, gn_b, yh);
  gemm_wmma<2><<<dim3(Cn / 16, Mn / 128, 1), blk, 0, stream>>>(yh, wph, d_out, bp, HVDn, Cn);
}
